// SyntaxAwareLSTM_65549790872113
// MI455X (gfx1250) — matrix operand order repaired and checked by reading, not compiled, not run on hardware
//
#include <hip/hip_runtime.h>
#include <hip/hip_bf16.h>

// ---------------- problem constants ----------------
#define SEQ   8192
#define INDIM 512
#define MDIM  512
#define G5    2560      // 5*M
#define KREL  8
#define NWG_DIR 11      // workgroups per direction in the recurrent kernel

typedef _Float16 v16h __attribute__((ext_vector_type(16)));
typedef _Float16 v8h  __attribute__((ext_vector_type(8)));
typedef float    v8f  __attribute__((ext_vector_type(8)));

__device__ __forceinline__ float sigm(float x) { return 1.0f / (1.0f + __expf(-x)); }

// ---------------- fp32 -> fp16 convert ----------------
__global__ void f32_to_f16_kernel(const float* __restrict__ src,
                                  _Float16* __restrict__ dst, int n) {
  int i = blockIdx.x * blockDim.x + threadIdx.x;
  if (i < n) dst[i] = (_Float16)src[i];
}

// ---------------- batched GEMM:  C[rows,N] = A[rows,512] * W[N,512]^T + bias ----------------
// f16 inputs, f32 accumulate via v_wmma_f32_16x16x32_f16.
// block = 256 threads (8 waves); each wave owns one 16x16 C tile; grid = (rows/16, N/128).
__global__ __launch_bounds__(256, 1)
void gemm_nt_wmma_kernel(const _Float16* __restrict__ A,
                         const _Float16* __restrict__ W,
                         const float* __restrict__ bias,
                         float* __restrict__ C, int N) {
  const int wave = threadIdx.x >> 5;
  const int lane = threadIdx.x & 31;
  const int lr   = lane & 15;   // row (A) / col (B,C) within tile
  const int lh   = lane >> 4;   // half selector
  const int stile = blockIdx.x;
  const int jtile = blockIdx.y * 8 + wave;
  const int j0 = jtile * 16;
  const int rowA = stile * 16 + lr;

  v8f acc = {};
  for (int kc = 0; kc < INDIM / 32; ++kc) {
    const int kb = kc * 32;
    // A fragment: lane lr/lh holds K = lh*16+(0..15), contiguous (matches B)
    const _Float16* ap = A + (size_t)rowA * INDIM + kb + lh * 16;
    v8h alo = *(const v8h*)(ap);
    v8h ahi = *(const v8h*)(ap + 8);
    v16h a;
#pragma unroll
    for (int i = 0; i < 8; ++i) { a[i] = alo[i]; a[8 + i] = ahi[i]; }
    // B fragment: column n = lr of B is row (j0+lr) of W, contiguous over K
    const _Float16* bp = W + (size_t)(j0 + lr) * INDIM + kb + lh * 16;
    v16h b = *(const v16h*)(bp);
    acc = __builtin_amdgcn_wmma_f32_16x16x32_f16(false, a, false, b,
                                                 (short)0, acc, false, false);
  }
  // C layout: lanes 0-15 -> M = v, lanes 16-31 -> M = v+8; N = lr
  const int col = j0 + lr;
  const float bv = bias[col];
#pragma unroll
  for (int v = 0; v < 8; ++v) {
    const int row = stile * 16 + v + 8 * lh;
    C[(size_t)row * N + col] = acc[v] + bv;
  }
}

// ---------------- device-wide barrier (monotonic counter) ----------------
__device__ __forceinline__ void grid_barrier(int* ctr, int nwg, int* ph) {
  __threadfence();
  __syncthreads();
  if (threadIdx.x == 0)
    __hip_atomic_fetch_add(ctr, 1, __ATOMIC_ACQ_REL, __HIP_MEMORY_SCOPE_AGENT);
  ++(*ph);
  if (threadIdx.x == 0) {
    const int target = (*ph) * nwg;
    while (__hip_atomic_load(ctr, __ATOMIC_ACQUIRE, __HIP_MEMORY_SCOPE_AGENT) < target)
      __builtin_amdgcn_s_sleep(8);
  }
  __syncthreads();
}

// ---------------- persistent recurrent kernel ----------------
// grid = 2*NWG_DIR blocks of 256. dir = blockIdx.x / NWG_DIR.
// WG 0..9 : ious[j] = Xiou[t,j] + biouh[j] + dot(Wiouh[j,:], h_prev)     (phase 1)
// WG 10   : relout[k,m] = dot(Wlh[m,:], rel_h[k,:]) via WMMA             (phase 1)
// WG 0..1 : gates + masked relative aggregation -> h, c, outputs         (phase 2)
__global__ __launch_bounds__(256, 1)
void syn_lstm_seq_kernel(const float* __restrict__ Xiou, const float* __restrict__ Xl,
                         const _Float16* __restrict__ Wiouh_h, const _Float16* __restrict__ Wlh_h,
                         const float* __restrict__ biouh, const float* __restrict__ blh,
                         const float* __restrict__ lb,
                         const int* __restrict__ rel_fw, const int* __restrict__ dep_fw,
                         const unsigned char* __restrict__ mask_fw,
                         const int* __restrict__ rel_bw, const int* __restrict__ dep_bw,
                         const unsigned char* __restrict__ mask_bw,
                         float* __restrict__ Hd,            // d_out [SEQ, 2*MDIM]
                         _Float16* __restrict__ Hh_base,    // [2][SEQ*MDIM] f16 mirror
                         float* __restrict__ iou_base,      // [2][G5]
                         float* __restrict__ relout_base,   // [2][KREL*MDIM]
                         float* __restrict__ cbuf_base,     // [2][MDIM]
                         int* __restrict__ bar_base) {
  const int tid = threadIdx.x;
  const int dir = blockIdx.x / NWG_DIR;
  const int wg  = blockIdx.x % NWG_DIR;
  const int diroff = dir * MDIM;

  const int* rel            = dir ? rel_bw  : rel_fw;
  const int* dep            = dir ? dep_bw  : dep_fw;
  const unsigned char* mask = dir ? mask_bw : mask_fw;
  _Float16* Hh   = Hh_base     + (size_t)dir * SEQ * MDIM;
  float* iouBuf  = iou_base    + dir * G5;
  float* relout  = relout_base + dir * (KREL * MDIM);
  float* cbuf    = cbuf_base   + dir * MDIM;
  int* ctr       = bar_base    + dir * 64;   // separate cache lines

  __shared__ float    hprev[MDIM];
  __shared__ _Float16 Alds[16 * MDIM];       // padded rel_h, 16KB

  int ph = 0;
  for (int s = 0; s < SEQ; ++s) {
    const int t = dir ? (SEQ - 1 - s) : s;
    const int tprev = dir ? (t + 1) : (t - 1);

    // ---------------- phase 1 ----------------
    if (wg < 10) {
      for (int m = tid; m < MDIM; m += 256)
        hprev[m] = (s == 0) ? 0.0f : Hd[(size_t)tprev * (2 * MDIM) + diroff + m];
      __syncthreads();
      const int j = wg * 256 + tid;
      const _Float16* wrow = Wiouh_h + (size_t)j * MDIM;
      float acc = Xiou[(size_t)t * G5 + j] + biouh[j];
      for (int k0 = 0; k0 < MDIM; k0 += 8) {
        v8h wv = *(const v8h*)(wrow + k0);
#pragma unroll
        for (int q = 0; q < 8; ++q) acc += (float)wv[q] * hprev[k0 + q];
      }
      iouBuf[j] = acc;
    } else {
      // stage padded A = [rel_h(8 rows); zeros(8 rows)] in LDS (f16)
      for (int idx = tid; idx < KREL * MDIM; idx += 256) {
        const int k = idx >> 9, off = idx & (MDIM - 1);
        const int r = rel[t * KREL + k];
        Alds[idx] = Hh[(size_t)r * MDIM + off];     // garbage rows are masked later
      }
      for (int idx = tid; idx < KREL * MDIM; idx += 256)
        Alds[KREL * MDIM + idx] = (_Float16)0.0f;
      __syncthreads();

      const int wave = tid >> 5, lane = tid & 31;
      const int lr = lane & 15, lh = lane >> 4;
      for (int q = 0; q < 4; ++q) {
        const int j0 = (wave * 4 + q) * 16;
        v8f acc = {};
        for (int kc = 0; kc < MDIM / 32; ++kc) {
          const int kb = kc * 32;
          const _Float16* ap = &Alds[lr * MDIM + kb + lh * 16];
          v8h alo = *(const v8h*)(ap);
          v8h ahi = *(const v8h*)(ap + 8);
          v16h a;
#pragma unroll
          for (int i = 0; i < 8; ++i) { a[i] = alo[i]; a[8 + i] = ahi[i]; }
          const _Float16* bp = Wlh_h + (size_t)(j0 + lr) * MDIM + kb + lh * 16;
          v16h b = *(const v16h*)(bp);
          acc = __builtin_amdgcn_wmma_f32_16x16x32_f16(false, a, false, b,
                                                       (short)0, acc, false, false);
        }
        if (lh == 0) {   // lanes 0-15 hold C rows 0..7 == the 8 real relatives
#pragma unroll
          for (int v = 0; v < 8; ++v) relout[v * MDIM + j0 + lr] = acc[v];
        }
      }
    }
    grid_barrier(ctr, NWG_DIR, &ph);

    // ---------------- phase 2 ----------------
    if (wg < 2) {
      const int m = wg * 256 + tid;
      const float gi = sigm(iouBuf[0 * MDIM + m]);
      const float go = sigm(iouBuf[1 * MDIM + m]);
      const float gs = sigm(iouBuf[2 * MDIM + m]);   // chunk 2 -> s (faithful swap)
      const float gf = sigm(iouBuf[3 * MDIM + m]);   // chunk 3 -> f
      const float gu = tanhf(iouBuf[4 * MDIM + m]);
      const float cold = (s == 0) ? 0.0f : cbuf[m];
      const float cnew = gf * cold + gi * gu;
      float h = go * tanhf(cnew);
      const float xlm = Xl[(size_t)t * MDIM + m] + blh[m];
      float agg = 0.0f;
#pragma unroll
      for (int k = 0; k < KREL; ++k) {
        if (mask[t * KREL + k]) {
          const int r = rel[t * KREL + k];
          const float rh = Hd[(size_t)r * (2 * MDIM) + diroff + m];
          const float lv = sigm(relout[k * MDIM + m] + xlm +
                                lb[(size_t)dep[t * KREL + k] * MDIM + m]);
          agg += lv * rh;
        }
      }
      h += gs * tanhf(agg);
      cbuf[m] = cnew;
      Hd[(size_t)t * (2 * MDIM) + diroff + m] = h;
      Hh[(size_t)t * MDIM + m] = (_Float16)h;
    }
    grid_barrier(ctr, NWG_DIR, &ph);
  }
}

// ---------------- launch ----------------
extern "C" void kernel_launch(void* const* d_in, const int* in_sizes, int n_in,
                              void* d_out, int out_size, void* d_ws, size_t ws_size,
                              hipStream_t stream) {
  const float* inputs = (const float*)d_in[0];
  const float* Wioux  = (const float*)d_in[1];
  const float* bioux  = (const float*)d_in[2];
  const float* Wiouh  = (const float*)d_in[3];
  const float* biouh  = (const float*)d_in[4];
  const float* Wlx    = (const float*)d_in[5];
  const float* blx    = (const float*)d_in[6];
  const float* Wlh    = (const float*)d_in[7];
  const float* blh    = (const float*)d_in[8];
  const float* lb     = (const float*)d_in[9];
  const int* rel_fw   = (const int*)d_in[10];
  const int* dep_fw   = (const int*)d_in[11];
  const unsigned char* mask_fw = (const unsigned char*)d_in[12];
  const int* rel_bw   = (const int*)d_in[13];
  const int* dep_bw   = (const int*)d_in[14];
  const unsigned char* mask_bw = (const unsigned char*)d_in[15];

  char* ws = (char*)d_ws;
  size_t off = 0;
  auto carve = [&](size_t bytes) -> char* {
    char* p = ws + off;
    off = (off + bytes + 255) & ~(size_t)255;
    return p;
  };
  int*       bar      = (int*)      carve(2 * 64 * sizeof(int));
  _Float16*  Ah       = (_Float16*) carve((size_t)SEQ * INDIM * 2);
  _Float16*  Wioux_h  = (_Float16*) carve((size_t)G5 * INDIM * 2);
  _Float16*  Wiouh_h  = (_Float16*) carve((size_t)G5 * INDIM * 2);
  _Float16*  Wlx_h    = (_Float16*) carve((size_t)MDIM * INDIM * 2);
  _Float16*  Wlh_h    = (_Float16*) carve((size_t)MDIM * INDIM * 2);
  _Float16*  Hh       = (_Float16*) carve((size_t)2 * SEQ * MDIM * 2);
  float*     Xiou     = (float*)    carve((size_t)SEQ * G5 * 4);
  float*     Xl       = (float*)    carve((size_t)SEQ * MDIM * 4);
  float*     iouBuf   = (float*)    carve((size_t)2 * G5 * 4);
  float*     relout   = (float*)    carve((size_t)2 * KREL * MDIM * 4);
  float*     cbuf     = (float*)    carve((size_t)2 * MDIM * 4);

  // zero the barrier counters (graph-capturable)
  hipMemsetAsync(bar, 0, 2 * 64 * sizeof(int), stream);

  // fp32 -> fp16 mirrors
  auto cvt = [&](const float* s, _Float16* d, int n) {
    f32_to_f16_kernel<<<(n + 255) / 256, 256, 0, stream>>>(s, d, n);
  };
  cvt(inputs, Ah,      SEQ * INDIM);
  cvt(Wioux,  Wioux_h, G5 * INDIM);
  cvt(Wiouh,  Wiouh_h, G5 * INDIM);
  cvt(Wlx,    Wlx_h,   MDIM * INDIM);
  cvt(Wlh,    Wlh_h,   MDIM * INDIM);

  // hoisted x-projections (WMMA GEMMs)
  gemm_nt_wmma_kernel<<<dim3(SEQ / 16, G5 / 128),   256, 0, stream>>>(Ah, Wioux_h, bioux, Xiou, G5);
  gemm_nt_wmma_kernel<<<dim3(SEQ / 16, MDIM / 128), 256, 0, stream>>>(Ah, Wlx_h,   blx,   Xl,   MDIM);

  // persistent bidirectional recurrence
  syn_lstm_seq_kernel<<<2 * NWG_DIR, 256, 0, stream>>>(
      Xiou, Xl, Wiouh_h, Wlh_h, biouh, blh, lb,
      rel_fw, dep_fw, mask_fw, rel_bw, dep_bw, mask_bw,
      (float*)d_out, Hh, iouBuf, relout, cbuf, bar);
}
